// CNNRNN_33139967656429
// MI455X (gfx1250) — matrix operand order repaired and checked by reading, not compiled, not run on hardware
//
#include <hip/hip_runtime.h>
#include <hip/hip_bf16.h>
#include <math.h>

// ---------------- model constants ----------------
#define BATCH   8
#define TSTEPS  20
#define EDIM    22
#define VDIM    22
#define HLSTM   512
#define STARTTK 20

// ---------------- WMMA vector types ----------------
typedef __bf16 v16bf __attribute__((ext_vector_type(16)));
typedef __bf16 v8bf  __attribute__((ext_vector_type(8)));
typedef __bf16 v4bf  __attribute__((ext_vector_type(4)));
typedef float  v8f   __attribute__((ext_vector_type(8)));

// ---------------- fp32 -> bf16 bulk convert (weights/images, once/launch) --
__global__ void f32_to_bf16_kernel(const float* __restrict__ src,
                                   __bf16* __restrict__ dst, int n)
{
    int i = (blockIdx.x * blockDim.x + threadIdx.x) * 4;
    if (i + 3 < n) {
        float4 v = *(const float4*)(src + i);
        v4bf t;
        t[0] = (__bf16)v.x; t[1] = (__bf16)v.y;
        t[2] = (__bf16)v.z; t[3] = (__bf16)v.w;
        *(v4bf*)&dst[i] = t;
    } else {
        for (int j = i; j < n; ++j) dst[j] = (__bf16)src[j];
    }
}

// =====================================================================
// 3x3 SAME conv + bias + ReLU, implicit GEMM via WMMA bf16.
//   All tensors bf16 (f32 accumulate). Block = 128 thr (4 waves).
//   Block tile: 64 Cout x 64 spatial. Wave: 16 Cout x 4 N-subtiles
//   -> 4 WMMAs / K-step / wave off one A fragment + one staged B tile.
//   Software-pipelined with the LAST K-step peeled: steady-state body is
//   branch-free (barrier -> ds_loads -> next gather/A-load -> 4 WMMA ->
//   ds_store rotate). Double-buffered LDS B tile; one barrier per step.
// =====================================================================
template<bool FULLK>
__global__ __launch_bounds__(128) void conv3x3_wmma_kernel(
    const __bf16* __restrict__ in,   // [B, Cin, H, W]
    const __bf16* __restrict__ wgt,  // [Cout, Cin, 3, 3] (preconverted)
    const float*  __restrict__ bias, // [Cout]
    __bf16* __restrict__ out,        // [B, Cout, H, W]
    int Cin, int Cout, int H, int W, int S /* BATCH*H*W */)
{
    const int HW   = H * W;
    const int K    = Cin * 9;
    const int lane = threadIdx.x & 31;
    const int wave = threadIdx.x >> 5;
    const int tid  = threadIdx.x;
    const int sBase = blockIdx.x * 64;                 // spatial tile base
    const int m0    = blockIdx.y * 64 + wave * 16;     // cout base for wave

    __shared__ __align__(32) __bf16 ldsB[2][64 * 32];  // [buf][n][kk]

    // ---- staging: thread owns spatial n (0..63) and kk half (0..1) ----
    const int n_st  = tid & 63;
    const int kk0   = (tid >> 6) << 4;     // 0 or 16
    const int sg_st = sBase + n_st;
    const bool sOK  = sg_st < S;
    const int sg_c  = sOK ? sg_st : 0;
    const int b_st  = sg_c / HW;
    const int hw_st = sg_c - b_st * HW;
    const int h_st  = hw_st / W;
    const int w_st  = hw_st - h_st * W;
    const size_t inbase = (size_t)b_st * Cin * HW;

    // incremental (ci, r) for k = kb + kk0 ; advances by 32 per K-step
    int ci = (kk0 == 16) ? 1 : 0;
    int r  = (kk0 == 16) ? 7 : 0;

    const int col = lane & 15;
    const int hi  = lane >> 4;
    const int kSteps = (K + 31) >> 5;

    // gather one K-step of the input patch into registers (advances ci/r)
    auto gather = [&](int ks, v8bf& t0, v8bf& t1) {
        const int kb = ks << 5;
        int cij = ci, rj = r;
        #pragma unroll
        for (int j = 0; j < 16; ++j) {
            int kh = (rj >= 6) ? 2 : (rj >= 3 ? 1 : 0);
            int kw = rj - 3 * kh;
            int ih = h_st + kh - 1, iw = w_st + kw - 1;
            bool ok = sOK & (ih >= 0) & (ih < H) & (iw >= 0) & (iw < W);
            if (!FULLK) ok = ok & ((kb + kk0 + j) < K);
            __bf16 v = ok ? in[inbase + (size_t)cij * HW + ih * W + iw]
                          : (__bf16)0.f;
            if (j < 8) t0[j] = v; else t1[j - 8] = v;
            rj += 1; if (rj == 9) { rj = 0; cij += 1; }
        }
        ci += 3; r += 5; if (r >= 9) { r -= 9; ci += 1; }
    };
    // A fragment (bf16 weights): two contiguous 16B loads
    auto loadA = [&](int ks) -> v16bf {
        const int kb = ks << 5;
        v16bf a;
        if (FULLK) {
            const __bf16* wrow = wgt + (size_t)(m0 + col) * K + kb + (hi << 4);
            v8bf lo = *(const v8bf*)(wrow);
            v8bf hh = *(const v8bf*)(wrow + 8);
            #pragma unroll
            for (int e = 0; e < 8; ++e) { a[e] = lo[e]; a[e + 8] = hh[e]; }
        } else {
            #pragma unroll
            for (int e = 0; e < 16; ++e) {
                int k = kb + (hi << 4) + e;
                a[e] = (k < K) ? wgt[(size_t)(m0 + col) * K + k] : (__bf16)0.f;
            }
        }
        return a;
    };

    v8f acc0 = {}, acc1 = {}, acc2 = {}, acc3 = {};

    // ---- prologue: stage K-step 0 into buffer 0 ----
    {
        v8bf t0, t1;
        gather(0, t0, t1);
        *(v8bf*)&ldsB[0][n_st * 32 + kk0]     = t0;
        *(v8bf*)&ldsB[0][n_st * 32 + kk0 + 8] = t1;
    }
    v16bf aCur = loadA(0);

    // ---- steady state: branch-free body, last K-step peeled ----
    int ks = 0;
    for (; ks < kSteps - 1; ++ks) {
        const int buf = ks & 1;
        __syncthreads();                // buf's stores (prev iter) visible

        v16bf bm0 = *(const v16bf*)&ldsB[buf][(col)      * 32 + (hi << 4)];
        v16bf bm1 = *(const v16bf*)&ldsB[buf][(col + 16) * 32 + (hi << 4)];
        v16bf bm2 = *(const v16bf*)&ldsB[buf][(col + 32) * 32 + (hi << 4)];
        v16bf bm3 = *(const v16bf*)&ldsB[buf][(col + 48) * 32 + (hi << 4)];

        // issue next step's global loads BEFORE the matrix ops
        v8bf t0, t1;
        gather(ks + 1, t0, t1);
        v16bf aNext = loadA(ks + 1);
        if (FULLK && (ks + 2) < kSteps)
            __builtin_prefetch(
                wgt + (size_t)(m0 + col) * K + ((ks + 2) << 5) + (hi << 3),
                0, 1);

        acc0 = __builtin_amdgcn_wmma_f32_16x16x32_bf16(false, aCur, false, bm0, (short)0, acc0, false, false);
        acc1 = __builtin_amdgcn_wmma_f32_16x16x32_bf16(false, aCur, false, bm1, (short)0, acc1, false, false);
        acc2 = __builtin_amdgcn_wmma_f32_16x16x32_bf16(false, aCur, false, bm2, (short)0, acc2, false, false);
        acc3 = __builtin_amdgcn_wmma_f32_16x16x32_bf16(false, aCur, false, bm3, (short)0, acc3, false, false);

        // rotate: store next tile into the other buffer (loadcnt wait here)
        *(v8bf*)&ldsB[buf ^ 1][n_st * 32 + kk0]     = t0;
        *(v8bf*)&ldsB[buf ^ 1][n_st * 32 + kk0 + 8] = t1;
        aCur = aNext;
    }
    // ---- final K-step (no staging) ----
    {
        const int buf = ks & 1;
        __syncthreads();
        v16bf bm0 = *(const v16bf*)&ldsB[buf][(col)      * 32 + (hi << 4)];
        v16bf bm1 = *(const v16bf*)&ldsB[buf][(col + 16) * 32 + (hi << 4)];
        v16bf bm2 = *(const v16bf*)&ldsB[buf][(col + 32) * 32 + (hi << 4)];
        v16bf bm3 = *(const v16bf*)&ldsB[buf][(col + 48) * 32 + (hi << 4)];
        acc0 = __builtin_amdgcn_wmma_f32_16x16x32_bf16(false, aCur, false, bm0, (short)0, acc0, false, false);
        acc1 = __builtin_amdgcn_wmma_f32_16x16x32_bf16(false, aCur, false, bm1, (short)0, acc1, false, false);
        acc2 = __builtin_amdgcn_wmma_f32_16x16x32_bf16(false, aCur, false, bm2, (short)0, acc2, false, false);
        acc3 = __builtin_amdgcn_wmma_f32_16x16x32_bf16(false, aCur, false, bm3, (short)0, acc3, false, false);
    }

    // ---- store D: N = lane%16 (+16t), M = rr + 8*hi; + bias, ReLU, bf16 ----
    #pragma unroll
    for (int t = 0; t < 4; ++t) {
        int sg = sBase + t * 16 + col;
        if (sg >= S) continue;
        int b  = sg / HW;
        int hw = sg - b * HW;
        const v8f& acc = (t == 0) ? acc0 : (t == 1) ? acc1 : (t == 2) ? acc2 : acc3;
        #pragma unroll
        for (int rr = 0; rr < 8; ++rr) {
            int co = m0 + (hi << 3) + rr;
            float v = acc[rr] + bias[co];
            out[((size_t)b * Cout + co) * HW + hw] = (__bf16)(v > 0.f ? v : 0.f);
        }
    }
}

// =====================================================================
// GEMM via WMMA bf16: out[m][n] = act(A[m][:] . Wt[n][:] + bias[n])
//   A: bf16 [Mrows, K] (Mrows <= 16, zero-padded), Wt: fp32 [N, K]
//   (FC weights read exactly once -> direct fp32 read + HW cvt is the
//   minimum-traffic choice). K % 32 == 0, K >= 64. Output bf16 or f32.
//   Same peeled software pipeline as the conv kernel.
// =====================================================================
__global__ __launch_bounds__(128) void gemm_wmma_kernel(
    const __bf16* __restrict__ A,
    const float*  __restrict__ Wt,
    const float*  __restrict__ bias,
    __bf16* __restrict__ outB,       // used if outF == nullptr
    float*  __restrict__ outF,
    int K, int N, int Mrows, int doRelu)
{
    const int lane = threadIdx.x & 31;
    const int wave = threadIdx.x >> 5;
    const int tid  = threadIdx.x;
    const int n0   = blockIdx.x * 64 + wave * 16;

    __shared__ __align__(32) __bf16 ldsA[2][16 * 32];    // [buf][m][kk]

    const int m_st  = tid >> 3;         // 0..15
    const int kk0   = (tid & 7) * 4;    // 0..28

    const int col = lane & 15, hi = lane >> 4;
    const int ncol   = n0 + col;
    const bool nok   = ncol < N;
    const size_t wro = (size_t)(nok ? ncol : 0) * K;

    auto stageA = [&](int ks) -> v4bf {
        v4bf t = {};
        if (m_st < Mrows)
            t = *(const v4bf*)(A + (size_t)m_st * K + (ks << 5) + kk0);
        return t;
    };
    auto loadB = [&](int ks) -> v16bf {
        float wv[16];
        if (nok) {
            const float* pw = Wt + wro + (ks << 5) + (hi << 4);
            #pragma unroll
            for (int q = 0; q < 4; ++q)
                *(float4*)&wv[q * 4] = *(const float4*)(pw + q * 4);
        } else {
            #pragma unroll
            for (int q = 0; q < 16; ++q) wv[q] = 0.f;
        }
        v16bf bv;
        #pragma unroll
        for (int e = 0; e < 16; ++e) bv[e] = (__bf16)wv[e];
        return bv;
    };
    auto fragA = [&](int buf) -> v16bf {
        v8bf lo = *(const v8bf*)&ldsA[buf][col * 32 + (hi << 4)];
        v8bf hh = *(const v8bf*)&ldsA[buf][col * 32 + (hi << 4) + 8];
        v16bf av;
        #pragma unroll
        for (int e = 0; e < 8; ++e) { av[e] = lo[e]; av[e + 8] = hh[e]; }
        return av;
    };

    v8f acc = {};
    const int kSteps = K >> 5;

    {
        v4bf t = stageA(0);
        *(v4bf*)&ldsA[0][m_st * 32 + kk0] = t;
    }
    v16bf bCur = loadB(0);

    int ks = 0;
    for (; ks < kSteps - 1; ++ks) {
        const int buf = ks & 1;
        __syncthreads();
        v16bf av = fragA(buf);
        v4bf tn = stageA(ks + 1);
        v16bf bNext = loadB(ks + 1);

        acc = __builtin_amdgcn_wmma_f32_16x16x32_bf16(
                  false, av, false, bCur, (short)0, acc, false, false);

        *(v4bf*)&ldsA[buf ^ 1][m_st * 32 + kk0] = tn;
        bCur = bNext;
    }
    {
        const int buf = ks & 1;
        __syncthreads();
        v16bf av = fragA(buf);
        acc = __builtin_amdgcn_wmma_f32_16x16x32_bf16(
                  false, av, false, bCur, (short)0, acc, false, false);
    }

    if (nok) {
        float bb = bias[ncol];
        #pragma unroll
        for (int rr = 0; rr < 8; ++rr) {
            int m = (hi << 3) + rr;
            if (m < Mrows) {
                float v = acc[rr] + bb;
                if (doRelu) v = v > 0.f ? v : 0.f;
                if (outF) outF[(size_t)m * N + ncol] = v;
                else      outB[(size_t)m * N + ncol] = (__bf16)v;
            }
        }
    }
}

// ---------------- 2x2 max pool, stride 2, NCHW, bf16 ----------------
__global__ void maxpool2_kernel(const __bf16* __restrict__ in,
                                __bf16* __restrict__ out,
                                int C, int H, int W, int total)
{
    int i = blockIdx.x * blockDim.x + threadIdx.x;
    if (i >= total) return;
    int Wo = W >> 1, Ho = H >> 1;
    int wo = i % Wo; int t = i / Wo;
    int ho = t % Ho; t /= Ho;
    int c  = t % C;  int b = t / C;
    const __bf16* p = in + ((size_t)(b * C + c) * H + ho * 2) * W + wo * 2;
    float m = fmaxf(fmaxf((float)p[0], (float)p[1]),
                    fmaxf((float)p[W], (float)p[W + 1]));
    out[i] = (__bf16)m;
}

// ---------------- zero-fill ----------------
__global__ void zero_fill_kernel(float* p, int n)
{
    int i = blockIdx.x * blockDim.x + threadIdx.x;
    if (i < n) p[i] = 0.f;
}

// ---------------- embedding of teacher-forced tokens ----------------
__global__ void embed_kernel(const int* __restrict__ tokens,
                             const float* __restrict__ embed_W,
                             float* __restrict__ emb /* [B,T,E] */)
{
    int i = blockIdx.x * blockDim.x + threadIdx.x;
    if (i >= BATCH * TSTEPS * EDIM) return;
    int e = i % EDIM;
    int t = (i / EDIM) % TSTEPS;
    int b = i / (EDIM * TSTEPS);
    int tok = (t == 0) ? STARTTK : tokens[b * TSTEPS + (t - 1)];
    emb[i] = embed_W[tok * EDIM + e];
}

// ---------------- one LSTM step (fp32; 0.15% of total FLOPs) ----------------
__global__ void lstm_step_kernel(const float* __restrict__ emb, int t,
                                 const float* __restrict__ hprev,
                                 const float* __restrict__ cprev,
                                 float* __restrict__ hnext,
                                 float* __restrict__ cnext,
                                 float* __restrict__ hs /* [T,B,H] */,
                                 const float* __restrict__ w_ih,
                                 const float* __restrict__ w_hh,
                                 const float* __restrict__ b_ih,
                                 const float* __restrict__ b_hh)
{
    int i = blockIdx.x * blockDim.x + threadIdx.x;   // 0 .. B*H-1
    if (i >= BATCH * HLSTM) return;
    int j = i % HLSTM;
    int b = i / HLSTM;
    const float* e = emb + ((size_t)b * TSTEPS + t) * EDIM;
    const float* h = hprev + (size_t)b * HLSTM;

    float g[4];
    #pragma unroll
    for (int q = 0; q < 4; ++q) {
        int row = q * HLSTM + j;
        float s = b_ih[row] + b_hh[row];
        const float* wi = w_ih + (size_t)row * EDIM;
        #pragma unroll
        for (int k = 0; k < EDIM; ++k) s += e[k] * wi[k];
        const float* wh = w_hh + (size_t)row * HLSTM;
        for (int k = 0; k < HLSTM; ++k) s += h[k] * wh[k];
        g[q] = s;
    }
    float ig = 1.f / (1.f + __expf(-g[0]));
    float fg = 1.f / (1.f + __expf(-g[1]));
    float gg = tanhf(g[2]);
    float og = 1.f / (1.f + __expf(-g[3]));
    float c  = fg * cprev[i] + ig * gg;
    float hn = og * tanhf(c);
    cnext[i] = c;
    hnext[i] = hn;
    hs[((size_t)t * BATCH + b) * HLSTM + j] = hn;
}

// ---------------- xdec = relu(hs @ proj_rnn_w^T + proj_rnn_b + img_emb) ----
__global__ void xdec_kernel(const float* __restrict__ hs,
                            const float* __restrict__ proj_w, /* [E,H] */
                            const float* __restrict__ proj_b,
                            const float* __restrict__ img_emb, /* [B,E] */
                            float* __restrict__ xdec /* [T,B,E] */)
{
    int i = blockIdx.x * blockDim.x + threadIdx.x;
    if (i >= TSTEPS * BATCH * EDIM) return;
    int e = i % EDIM;
    int b = (i / EDIM) % BATCH;
    int t = i / (EDIM * BATCH);
    const float* h = hs + ((size_t)t * BATCH + b) * HLSTM;
    const float* w = proj_w + (size_t)e * HLSTM;
    float s = proj_b[e] + img_emb[b * EDIM + e];
    for (int k = 0; k < HLSTM; ++k) s += h[k] * w[k];
    xdec[i] = s > 0.f ? s : 0.f;
}

// ---------------- masked scores + log_softmax, one wave per (t,b) ----------
__global__ void logits_kernel(const float* __restrict__ xdec, /* [T,B,E] */
                              const float* __restrict__ embed_W, /* [V,E] */
                              float* __restrict__ out /* [B,T,V] */)
{
    int tb = blockIdx.x;          // 0 .. T*B-1
    int t  = tb / BATCH, b = tb % BATCH;
    int v  = threadIdx.x;         // 0..31 (one wave)
    const float* x = xdec + (size_t)tb * EDIM;

    float s = -3.402823e38f;
    if (v < VDIM) {
        float acc = 0.f;
        for (int e = 0; e < EDIM; ++e) {
            float w = (v == STARTTK) ? -1e9f : embed_W[v * EDIM + e];
            acc += x[e] * w;
        }
        s = acc;
    }
    float m = s;
    #pragma unroll
    for (int off = 16; off; off >>= 1) m = fmaxf(m, __shfl_xor(m, off, 32));
    float ex = (v < VDIM) ? __expf(s - m) : 0.f;
    float sum = ex;
    #pragma unroll
    for (int off = 16; off; off >>= 1) sum += __shfl_xor(sum, off, 32);
    if (v < VDIM)
        out[((size_t)b * TSTEPS + t) * VDIM + v] = s - m - logf(sum);
}

// =====================================================================
// host-side orchestration
// =====================================================================
extern "C" void kernel_launch(void* const* d_in, const int* in_sizes, int n_in,
                              void* d_out, int out_size, void* d_ws, size_t ws_size,
                              hipStream_t stream) {
    (void)in_sizes; (void)n_in; (void)out_size; (void)ws_size;

    static const int CIN[13]  = {3,64,64,128,128,256,256,256,512,512,512,512,512};
    static const int COUT[13] = {64,64,128,128,256,256,256,512,512,512,512,512,512};

    const float* cw[13]; const float* cb[13];
    for (int i = 0; i < 13; ++i) {
        cw[i] = (const float*)d_in[2 * i];
        cb[i] = (const float*)d_in[2 * i + 1];
    }
    const float* fc1_w      = (const float*)d_in[26];
    const float* fc1_b      = (const float*)d_in[27];
    const float* fc2_w      = (const float*)d_in[28];
    const float* fc2_b      = (const float*)d_in[29];
    const float* embed_W    = (const float*)d_in[30];
    const float* w_ih       = (const float*)d_in[31];
    const float* w_hh       = (const float*)d_in[32];
    const float* b_ih       = (const float*)d_in[33];
    const float* b_hh       = (const float*)d_in[34];
    const float* proj_rnn_w = (const float*)d_in[35];
    const float* proj_rnn_b = (const float*)d_in[36];
    const float* proj_img_w = (const float*)d_in[37];
    const float* proj_img_b = (const float*)d_in[38];
    const float* images     = (const float*)d_in[39];
    const int*   tokens     = (const int*)d_in[40];
    float* out = (float*)d_out;

    // ---- workspace carve-up: bf16 region first, then f32 region ----
    const size_t BIG = (size_t)BATCH * 64 * 224 * 224;   // 25,690,112 elems
    __bf16* bufA = (__bf16*)d_ws;
    __bf16* bufB = bufA + BIG;
    __bf16* wreg = bufB + BIG;                 // converted conv weights
    __bf16* cwb[13];
    {
        __bf16* q = wreg;
        for (int i = 0; i < 13; ++i) { cwb[i] = q; q += (size_t)COUT[i] * CIN[i] * 9; }
        wreg = q;
    }
    __bf16* imgbf = wreg;                      // bf16 images
    __bf16* fc1o  = imgbf + (size_t)BATCH * 3 * 224 * 224;
    __bf16* fc2o  = fc1o + (size_t)BATCH * 4096;
    float*  fp    = (float*)(fc2o + (size_t)BATCH * 4096);
    float* imge = fp;                fp += (size_t)BATCH * EDIM;
    float* emb  = fp;                fp += (size_t)BATCH * TSTEPS * EDIM;
    float* h0   = fp;                fp += (size_t)BATCH * HLSTM;
    float* c0   = fp;                fp += (size_t)BATCH * HLSTM;
    float* h1   = fp;                fp += (size_t)BATCH * HLSTM;
    float* c1   = fp;                fp += (size_t)BATCH * HLSTM;
    float* hs   = fp;                fp += (size_t)TSTEPS * BATCH * HLSTM;
    float* xd   = fp;                fp += (size_t)TSTEPS * BATCH * EDIM;

    // ---- preconvert conv weights + images to bf16 (once per launch) ----
    auto cvt = [&](const float* s, __bf16* d, int n) {
        f32_to_bf16_kernel<<<(n / 4 + 255) / 256, 256, 0, stream>>>(s, d, n);
    };
    for (int i = 0; i < 13; ++i) cvt(cw[i], cwb[i], COUT[i] * CIN[i] * 9);
    cvt(images, imgbf, BATCH * 3 * 224 * 224);

    auto conv = [&](const __bf16* in, int li, __bf16* o, int H, int W) {
        int Cin = CIN[li], Cout = COUT[li];
        int S = BATCH * H * W;
        dim3 grid((S + 63) / 64, Cout / 64);
        if ((Cin * 9) % 32 == 0)
            conv3x3_wmma_kernel<true><<<grid, 128, 0, stream>>>(in, cwb[li], cb[li], o, Cin, Cout, H, W, S);
        else
            conv3x3_wmma_kernel<false><<<grid, 128, 0, stream>>>(in, cwb[li], cb[li], o, Cin, Cout, H, W, S);
    };
    auto pool = [&](const __bf16* in, __bf16* o, int C, int H, int W) {
        int total = BATCH * C * (H / 2) * (W / 2);
        maxpool2_kernel<<<(total + 255) / 256, 256, 0, stream>>>(in, o, C, H, W, total);
    };

    // ---- VGG16 feature extractor (all bf16 activations) ----
    conv(imgbf, 0, bufA, 224, 224);
    conv(bufA,  1, bufB, 224, 224);
    pool(bufB, bufA, 64, 224, 224);                    // 64 x112x112
    conv(bufA,  2, bufB, 112, 112);
    conv(bufB,  3, bufA, 112, 112);
    pool(bufA, bufB, 128, 112, 112);                   // 128 x56x56
    conv(bufB,  4, bufA, 56, 56);
    conv(bufA,  5, bufB, 56, 56);
    conv(bufB,  6, bufA, 56, 56);
    pool(bufA, bufB, 256, 56, 56);                     // 256 x28x28
    conv(bufB,  7, bufA, 28, 28);
    conv(bufA,  8, bufB, 28, 28);
    conv(bufB,  9, bufA, 28, 28);
    pool(bufA, bufB, 512, 28, 28);                     // 512 x14x14
    conv(bufB, 10, bufA, 14, 14);
    conv(bufA, 11, bufB, 14, 14);
    conv(bufB, 12, bufA, 14, 14);
    pool(bufA, bufB, 512, 14, 14);                     // 512 x7x7 -> f = bufB [8,25088] bf16

    // ---- classifier + image projection (WMMA GEMMs) ----
    gemm_wmma_kernel<<<dim3(4096 / 64), 128, 0, stream>>>(
        bufB, fc1_w, fc1_b, fc1o, nullptr, 25088, 4096, BATCH, 1);
    gemm_wmma_kernel<<<dim3(4096 / 64), 128, 0, stream>>>(
        fc1o, fc2_w, fc2_b, fc2o, nullptr, 4096, 4096, BATCH, 1);
    gemm_wmma_kernel<<<dim3(1), 128, 0, stream>>>(
        fc2o, proj_img_w, proj_img_b, nullptr, imge, 4096, EDIM, BATCH, 0);

    // ---- decoder (fp32; negligible FLOPs) ----
    {
        int n = BATCH * TSTEPS * EDIM;
        embed_kernel<<<(n + 255) / 256, 256, 0, stream>>>(tokens, embed_W, emb);
    }
    {
        int n = 2 * BATCH * HLSTM;   // h0 and c0 are contiguous
        zero_fill_kernel<<<(n + 255) / 256, 256, 0, stream>>>(h0, n);
    }
    for (int t = 0; t < TSTEPS; ++t) {
        const float* hp = (t & 1) ? h1 : h0;
        const float* cp = (t & 1) ? c1 : c0;
        float* hn = (t & 1) ? h0 : h1;
        float* cn = (t & 1) ? c0 : c1;
        lstm_step_kernel<<<(BATCH * HLSTM + 255) / 256, 256, 0, stream>>>(
            emb, t, hp, cp, hn, cn, hs, w_ih, w_hh, b_ih, b_hh);
    }
    {
        int n = TSTEPS * BATCH * EDIM;
        xdec_kernel<<<(n + 255) / 256, 256, 0, stream>>>(hs, proj_rnn_w, proj_rnn_b, imge, xd);
    }
    logits_kernel<<<TSTEPS * BATCH, 32, 0, stream>>>(xd, embed_W, out);
}
